// Model_82875688944064
// MI455X (gfx1250) — matrix operand order repaired and checked by reading, not compiled, not run on hardware
//
#include <hip/hip_runtime.h>

// MI455X / gfx1250: wave32, WMMA 16x16x32 f16 -> f32
typedef __attribute__((ext_vector_type(16))) _Float16 v16h;
typedef __attribute__((ext_vector_type(8)))  float    v8f;

#define B_GRAPHS   1024
#define DEG        16
#define HID        32

// Build a 16x32 f16 A-fragment from 32 consecutive f32 at `base` (chunk start).
// ISA layout: lanes 0-15 elems0-7 = K+0..7, elems8-15 = K+16..23;
//             lanes 16-31 elems0-7 = K+8..15, elems8-15 = K+24..31.
// Requires base + hf*8 to be 16B aligned (true at all call sites).
__device__ __forceinline__ v16h afrag32(const float* base, int hf)
{
    const float4* p0 = reinterpret_cast<const float4*>(base + hf * 8);
    const float4* p1 = reinterpret_cast<const float4*>(base + 16 + hf * 8);
    float4 r0 = p0[0], r1 = p0[1];
    float4 r2 = p1[0], r3 = p1[1];
    v16h a;
    a[0]=(_Float16)r0.x; a[1]=(_Float16)r0.y; a[2]=(_Float16)r0.z; a[3]=(_Float16)r0.w;
    a[4]=(_Float16)r1.x; a[5]=(_Float16)r1.y; a[6]=(_Float16)r1.z; a[7]=(_Float16)r1.w;
    a[8]=(_Float16)r2.x; a[9]=(_Float16)r2.y; a[10]=(_Float16)r2.z; a[11]=(_Float16)r2.w;
    a[12]=(_Float16)r3.x; a[13]=(_Float16)r3.y; a[14]=(_Float16)r3.z; a[15]=(_Float16)r3.w;
    return a;
}

// -------------------------------------------------------------------------
// Fully fused GCN branch: one workgroup (256 thr = 8 waves) per graph.
//   Sl = Xg @ Wa                         (WMMA, X from global)
//   Hl = relu(Ag @ Sl + ba)              (LDS gather-reduce, no atomics:
//                                         rows == repeat(arange, DEG))
//   Sl = Hl @ Wb                         (WMMA, A-fragments from LDS)
//   Hl = relu(Ag @ Sl + bb)
//   feat[g] = [mean(Hl), max(Hl)]        (per-wave partials + reduce)
// Everything between the X read and the 64-float feat write stays in LDS.
// LDS rows padded to stride 36 floats -> bank-conflict-free gathers and
// conflict-free 16-row float4 fragment loads.
// -------------------------------------------------------------------------
template<int NPER>
__global__ __launch_bounds__(256)
void fused_branch(const float* __restrict__ X, const int* __restrict__ cols,
                  const float* __restrict__ vals,
                  const float* __restrict__ Wa, const float* __restrict__ ba,
                  const float* __restrict__ Wb, const float* __restrict__ bb,
                  float* __restrict__ feat, int featOff)
{
    constexpr int K     = NPER;                  // layer-a inner dim
    constexpr int KP    = ((K + 31) / 32) * 32;
    constexpr int NCH   = KP / 32;
    constexpr int NFULL = K / 32;
    constexpr int NT    = (NPER + 15) / 16;      // row tiles (8 / 13)
    constexpr int NPAD  = NT * 16;               // padded rows (128 / 208)
    constexpr int SROW  = 36;                    // LDS row stride (floats)

    __shared__ _Float16 WaL[NCH * 32 * 32];      // B-frag packed, zero-padded
    __shared__ _Float16 WbL[32 * 32];
    __shared__ float    Sl[NPAD * SROW];
    __shared__ float    Hl[NPAD * SROW];
    __shared__ float    Psum[8 * 32];
    __shared__ float    Pmax[8 * 32];

    const int tid = threadIdx.x;
    const int g   = blockIdx.x;

    // ---- stage weights: B-fragment order, each lane's 32 f16 contiguous ----
    for (int idx = tid; idx < NCH * 32 * 32; idx += 256) {
        int c = idx >> 10;  int r = idx & 1023;
        int l = r >> 5;     int e = r & 31;
        int nt = e >> 4,    j = e & 15;
        int k = c * 32 + (j >> 3) * 16 + (l >> 4) * 8 + (j & 7);
        int n = nt * 16 + (l & 15);
        WaL[idx] = (k < K) ? (_Float16)Wa[k * 32 + n] : (_Float16)0.f;
    }
    for (int idx = tid; idx < 1024; idx += 256) {
        int l = idx >> 5;  int e = idx & 31;
        int nt = e >> 4,   j = e & 15;
        int k = (j >> 3) * 16 + (l >> 4) * 8 + (j & 7);
        int n = nt * 16 + (l & 15);
        WbL[idx] = (_Float16)Wb[k * 32 + n];
    }
    __syncthreads();

    const int wave = tid >> 5;
    const int lane = tid & 31;
    const int hf   = lane >> 4;
    const int m    = lane & 15;

    // ---- layer-a GEMM: Sl = Xg @ Wa ----
    for (int t = wave; t < NT; t += 8) {
        int rl = t * 16 + m;  if (rl > NPER - 1) rl = NPER - 1;  // pad rows dup
        const float* arow = X + ((size_t)g * NPER + rl) * (size_t)K;
        v8f c0 = {}, c1 = {};
        #pragma unroll
        for (int c = 0; c < NFULL; ++c) {
            v16h a = afrag32(arow + c * 32, hf);
            const v16h* wp = reinterpret_cast<const v16h*>(&WaL[(c * 32 + lane) * 32]);
            v16h b0 = wp[0], b1 = wp[1];
            c0 = __builtin_amdgcn_wmma_f32_16x16x32_f16(false, a, false, b0, (short)0, c0, false, false);
            c1 = __builtin_amdgcn_wmma_f32_16x16x32_f16(false, a, false, b1, (short)0, c1, false, false);
        }
        if constexpr (K % 32 != 0) {
            constexpr int kb = NFULL * 32;
            v16h a;
            #pragma unroll
            for (int i = 0; i < 8; ++i) {
                int k0 = kb + hf * 8 + i;
                int k1 = kb + 16 + hf * 8 + i;
                float v0 = arow[k0 < K ? k0 : K - 1];
                float v1 = arow[k1 < K ? k1 : K - 1];
                a[i]     = (k0 < K) ? (_Float16)v0 : (_Float16)0.f;
                a[i + 8] = (k1 < K) ? (_Float16)v1 : (_Float16)0.f;
            }
            const v16h* wp = reinterpret_cast<const v16h*>(&WaL[(NFULL * 32 + lane) * 32]);
            v16h b0 = wp[0], b1 = wp[1];
            c0 = __builtin_amdgcn_wmma_f32_16x16x32_f16(false, a, false, b0, (short)0, c0, false, false);
            c1 = __builtin_amdgcn_wmma_f32_16x16x32_f16(false, a, false, b1, (short)0, c1, false, false);
        }
        #pragma unroll
        for (int j = 0; j < 8; ++j) {
            int row = t * 16 + hf * 8 + j;
            Sl[row * SROW + m]      = c0[j];
            Sl[row * SROW + 16 + m] = c1[j];
        }
    }
    __syncthreads();

    // ---- layer-a aggregation: Hl = relu(Ag @ Sl + ba) ----
    {
        const float bv = ba[lane];
        for (int nd = wave; nd < NPER; nd += 8) {
            size_t e = ((size_t)g * NPER + nd) * DEG + (lane & 15);
            int   cj = cols[e] - g * NPER;      // block-diagonal -> local index
            float vj = vals[e];
            float acc = bv;
            #pragma unroll
            for (int j = 0; j < DEG; ++j) {
                int   c = __shfl(cj, j, 32);
                float v = __shfl(vj, j, 32);
                acc = fmaf(v, Sl[c * SROW + lane], acc);
            }
            Hl[nd * SROW + lane] = fmaxf(acc, 0.f);
        }
    }
    __syncthreads();

    // ---- layer-b GEMM: Sl = Hl @ Wb  (A-fragments from LDS, K=32) ----
    {
        const v16h* wbp = reinterpret_cast<const v16h*>(&WbL[lane * 32]);
        v16h wb0 = wbp[0], wb1 = wbp[1];
        for (int t = wave; t < NT; t += 8) {
            int rl = t * 16 + m;  if (rl > NPER - 1) rl = NPER - 1;
            v16h a = afrag32(&Hl[rl * SROW], hf);
            v8f c0 = {}, c1 = {};
            c0 = __builtin_amdgcn_wmma_f32_16x16x32_f16(false, a, false, wb0, (short)0, c0, false, false);
            c1 = __builtin_amdgcn_wmma_f32_16x16x32_f16(false, a, false, wb1, (short)0, c1, false, false);
            #pragma unroll
            for (int j = 0; j < 8; ++j) {
                int row = t * 16 + hf * 8 + j;
                Sl[row * SROW + m]      = c0[j];
                Sl[row * SROW + 16 + m] = c1[j];
            }
        }
    }
    __syncthreads();

    // ---- layer-b aggregation: Hl = relu(Ag @ Sl + bb) ----
    {
        const float bv = bb[lane];
        for (int nd = wave; nd < NPER; nd += 8) {
            size_t e = ((size_t)g * NPER + nd) * DEG + (lane & 15);
            int   cj = cols[e] - g * NPER;
            float vj = vals[e];
            float acc = bv;
            #pragma unroll
            for (int j = 0; j < DEG; ++j) {
                int   c = __shfl(cj, j, 32);
                float v = __shfl(vj, j, 32);
                acc = fmaf(v, Sl[c * SROW + lane], acc);
            }
            Hl[nd * SROW + lane] = fmaxf(acc, 0.f);
        }
    }
    __syncthreads();

    // ---- pooling: per-wave partials, then wave 0 reduces ----
    {
        float s = 0.f, mx = -3.402823466e38f;
        for (int nd = wave; nd < NPER; nd += 8) {
            float v = Hl[nd * SROW + lane];
            s += v;  mx = fmaxf(mx, v);
        }
        Psum[wave * 32 + lane] = s;
        Pmax[wave * 32 + lane] = mx;
    }
    __syncthreads();
    if (wave == 0) {
        float s = 0.f, mx = -3.402823466e38f;
        #pragma unroll
        for (int w = 0; w < 8; ++w) {
            s += Psum[w * 32 + lane];
            mx = fmaxf(mx, Pmax[w * 32 + lane]);
        }
        feat[(size_t)g * 128 + featOff + lane]      = s / (float)NPER;
        feat[(size_t)g * 128 + featOff + 32 + lane] = mx;
    }
}

// -------------------------------------------------------------------------
// Generic dense GEMM for the MLP head (same design as before):
// S[N, NCOLS] = X[N, K] @ W[K, NCOLS] (+bias, ReLU). N % 128 == 0.
// -------------------------------------------------------------------------
template<int NCOLS, int K>
__global__ __launch_bounds__(256)
void gemm_wmma(const float* __restrict__ X, const float* __restrict__ W,
               const float* __restrict__ bias, float* __restrict__ S,
               int relu)
{
    constexpr int KP  = ((K + 31) / 32) * 32;
    constexpr int NCH = KP / 32;
    __shared__ _Float16 Wl[NCH * 32 * NCOLS];

    const int tid = threadIdx.x;
    for (int idx = tid; idx < NCH * 32 * NCOLS; idx += 256) {
        int c = idx / (32 * NCOLS);
        int r = idx - c * (32 * NCOLS);
        int l = r / NCOLS;
        int e = r - l * NCOLS;
        int nt = e >> 4, j = e & 15;
        int k = c * 32 + (j >> 3) * 16 + (l >> 4) * 8 + (j & 7);
        int n = nt * 16 + (l & 15);
        Wl[idx] = (k < K) ? (_Float16)W[k * NCOLS + n] : (_Float16)0.f;
    }
    __syncthreads();

    const int wave = tid >> 5;
    const int lane = tid & 31;
    const int hf   = lane >> 4;
    const int m    = lane & 15;
    const int mbase = blockIdx.x * 128 + wave * 16;

    const float* __restrict__ arow = X + (size_t)(mbase + m) * (size_t)K;
    v8f c0 = {}, c1 = {};
    #pragma unroll
    for (int c = 0; c < K / 32; ++c) {
        v16h a = afrag32(arow + c * 32, hf);
        const v16h* wp = reinterpret_cast<const v16h*>(&Wl[(c * 32 + lane) * NCOLS]);
        v16h b0 = wp[0];
        c0 = __builtin_amdgcn_wmma_f32_16x16x32_f16(false, a, false, b0, (short)0, c0, false, false);
        if (NCOLS == 32) {
            v16h b1 = wp[1];
            c1 = __builtin_amdgcn_wmma_f32_16x16x32_f16(false, a, false, b1, (short)0, c1, false, false);
        }
    }

    #pragma unroll
    for (int j = 0; j < 8; ++j) {
        int row = mbase + hf * 8 + j;
        float v = c0[j];
        if (bias) v += bias[m];
        if (relu) v = fmaxf(v, 0.f);
        S[(size_t)row * NCOLS + m] = v;
        if (NCOLS == 32) {
            float u = c1[j];
            if (bias) u += bias[16 + m];
            if (relu) u = fmaxf(u, 0.f);
            S[(size_t)row * NCOLS + 16 + m] = u;
        }
    }
}

// -------------------------------------------------------------------------
// fc3: [1024,16] @ [16,2] + b  (tiny tail, VALU)
// -------------------------------------------------------------------------
__global__ __launch_bounds__(256)
void fc3_kernel(const float* __restrict__ h2, const float* __restrict__ W,
                const float* __restrict__ b, float* __restrict__ out)
{
    int idx = blockIdx.x * 256 + threadIdx.x;
    if (idx >= B_GRAPHS * 2) return;
    int r = idx >> 1, c = idx & 1;
    float acc = b[c];
    #pragma unroll
    for (int k = 0; k < 16; ++k)
        acc = fmaf(h2[r * 16 + k], W[k * 2 + c], acc);
    out[idx] = acc;
}

// -------------------------------------------------------------------------
extern "C" void kernel_launch(void* const* d_in, const int* in_sizes, int n_in,
                              void* d_out, int out_size, void* d_ws, size_t ws_size,
                              hipStream_t stream)
{
    (void)in_sizes; (void)n_in; (void)out_size; (void)ws_size;
    // setup_inputs() order
    const int*   cols1 = (const int*)  d_in[1];
    const float* vals1 = (const float*)d_in[2];
    const float* x1    = (const float*)d_in[3];
    const int*   cols2 = (const int*)  d_in[5];
    const float* vals2 = (const float*)d_in[6];
    const float* x2    = (const float*)d_in[7];
    const float* W1a = (const float*)d_in[8],  *b1a = (const float*)d_in[9];
    const float* W1b = (const float*)d_in[10], *b1b = (const float*)d_in[11];
    const float* W2a = (const float*)d_in[12], *b2a = (const float*)d_in[13];
    const float* W2b = (const float*)d_in[14], *b2b = (const float*)d_in[15];
    const float* fc1W = (const float*)d_in[16], *fc1b = (const float*)d_in[17];
    const float* fc2W = (const float*)d_in[18], *fc2b = (const float*)d_in[19];
    const float* fc3W = (const float*)d_in[20], *fc3b = (const float*)d_in[21];
    float* out = (float*)d_out;

    // workspace: feat [1024x128] + h1 [1024x32] + h2 [1024x16]  (<1 MB)
    char* ws = (char*)d_ws;
    float* feat = (float*)ws;
    float* h1   = (float*)(ws + (size_t)B_GRAPHS * 128 * 4);
    float* h2   = (float*)(ws + (size_t)B_GRAPHS * (128 + 32) * 4);

    // ---- fused GCN branches: one workgroup per graph ----
    fused_branch<116><<<B_GRAPHS, 256, 0, stream>>>(x1, cols1, vals1, W1a, b1a, W1b, b1b, feat, 0);
    fused_branch<200><<<B_GRAPHS, 256, 0, stream>>>(x2, cols2, vals2, W2a, b2a, W2b, b2b, feat, 64);

    // ---- MLP head ----
    gemm_wmma<32,128><<<B_GRAPHS / 128, 256, 0, stream>>>(feat, fc1W, fc1b, h1, 1);
    gemm_wmma<16,32> <<<B_GRAPHS / 128, 256, 0, stream>>>(h1,   fc2W, fc2b, h2, 1);
    fc3_kernel       <<<(B_GRAPHS * 2 + 255) / 256, 256, 0, stream>>>(h2, fc3W, fc3b, out);
}
